// GATClassification_78314433675488
// MI455X (gfx1250) — matrix operand order repaired and checked by reading, not compiled, not run on hardware
//
#include <hip/hip_runtime.h>
#include <hip/hip_bf16.h>

// ---------------- problem constants (match reference) ----------------
#define NN    50000      // nodes
#define EE    800000     // edges before self loops
#define ETOT  850000     // edges incl. self loops
#define FE    16         // edge attr size
#define NH    4          // heads
#define CC    32         // channels per head
#define FEATD 128        // NH*CC
#define KK    10         // clusters
#define KT    4          // K/32 for all GEMMs (K = 128)
#define NEG_SLOPE 0.2f
#define BN_EPS 1e-5f

typedef __attribute__((ext_vector_type(16))) __bf16 v16bf;
typedef __attribute__((ext_vector_type(8)))  float  v8f;

// ---------------- helpers ----------------
__device__ __forceinline__ unsigned short bfbits(float f) {
    unsigned u = __float_as_uint(f);
    u += 0x7FFFu + ((u >> 16) & 1u);          // round-to-nearest-even
    return (unsigned short)(u >> 16);
}

// monotonic uint key for float atomic-max (works for all finite values)
__device__ __forceinline__ unsigned fkey(float f) {
    unsigned u = __float_as_uint(f);
    return (u & 0x80000000u) ? ~u : (u | 0x80000000u);
}
__device__ __forceinline__ float fkey_inv(unsigned k) {
    unsigned u = (k & 0x80000000u) ? (k & 0x7FFFFFFFu) : ~k;
    return __uint_as_float(u);
}

// ---------------- generic zero fill ----------------
__global__ void fill_zero(float* __restrict__ p, long n) {
    long i = (long)blockIdx.x * blockDim.x + threadIdx.x;
    long st = (long)gridDim.x * blockDim.x;
    for (; i < n; i += st) p[i] = 0.0f;
}

// ---------------- pack A (activations) into bf16 WMMA A-fragment layout ----------------
// Fragment index = (rowTile*KT + kt)*32 + lane ; 16 bf16 (32B) per fragment.
// ISA 7.12.2 16-bit A 16x32: lane half selects K groups {0-7,16-23} / {8-15,24-31}.
__global__ __launch_bounds__(256)
void pack_a(const float* __restrict__ A, __bf16* __restrict__ Apack, int rowTiles) {
    int idx = blockIdx.x * blockDim.x + threadIdx.x;   // == fragment index
    if (idx >= rowTiles * KT * 32) return;
    int lane    = idx & 31;
    int kt      = (idx >> 5) & (KT - 1);
    int rowTile = idx >> 7;
    int lo16 = lane & 15, halfL = lane >> 4;
    const float* arow = A + (long)(rowTile * 16 + lo16) * FEATD + kt * 32;
    unsigned out8[8];
#pragma unroll
    for (int p = 0; p < 8; ++p) {
        int k = (p < 4) ? (halfL * 8 + 2 * p) : (16 + halfL * 8 + 2 * (p - 4));
        out8[p] = (unsigned)bfbits(arow[k]) | ((unsigned)bfbits(arow[k + 1]) << 16);
    }
    uint4* dst = (uint4*)(Apack + (long)idx * 16);
    uint4 d0 = {out8[0], out8[1], out8[2], out8[3]};
    uint4 d1 = {out8[4], out8[5], out8[6], out8[7]};
    dst[0] = d0;
    dst[1] = d1;
}

// ---------------- pack B (weights) into bf16 WMMA B-fragment layout ----------------
// Fragment index = (colTile*KT + kt)*32 + lane. Column clamp handled here (branch-free).
// ISA 7.12.2 16-bit B 32x16: lane half selects K groups {0-7,16-23} / {8-15,24-31} for column lane&15.
__global__ void pack_b(const float* __restrict__ Bm, __bf16* __restrict__ Bpack,
                       int Ncols, int colTiles) {
    int idx = blockIdx.x * blockDim.x + threadIdx.x;
    if (idx >= colTiles * KT * 32) return;
    int lane = idx & 31;
    int kt   = (idx >> 5) & (KT - 1);
    int ct   = idx >> 7;
    int lo16 = lane & 15, halfL = lane >> 4;
    int col = ct * 16 + lo16;
    bool valid = (col < Ncols);
    int cc = valid ? col : 0;                          // clamped index -> unconditional load
    unsigned out8[8];
#pragma unroll
    for (int p = 0; p < 8; ++p) {
        int k = kt * 32 + ((p < 4) ? (halfL * 8 + 2 * p) : (16 + halfL * 8 + 2 * (p - 4)));
        unsigned b0 = valid ? (unsigned)bfbits(Bm[(long)k * Ncols + cc]) : 0u;
        unsigned b1 = valid ? (unsigned)bfbits(Bm[(long)(k + 1) * Ncols + cc]) : 0u;
        out8[p] = b0 | (b1 << 16);
    }
    uint4* dst = (uint4*)(Bpack + (long)idx * 16);
    uint4 d0 = {out8[0], out8[1], out8[2], out8[3]};
    uint4 d1 = {out8[4], out8[5], out8[6], out8[7]};
    dst[0] = d0;
    dst[1] = d1;
}

// ---------------- WMMA GEMM on pre-packed fragments ----------------
// One wave per 16-row tile; A fragments (K=128) held in registers and reused
// across all column tiles. Per column tile: one clause of 8 b128 loads, then
// 4 back-to-back v_wmma. Bias pointer is mandatory (zero buffer if unused).
__device__ __forceinline__ v8f wmma_bf16(v16bf a, v16bf b, v8f c) {
    return __builtin_amdgcn_wmma_f32_16x16x32_bf16(false, a, false, b,
                                                   (short)0, c, false, false);
}

__global__ __launch_bounds__(256)
void gemm_wmma_packed(const __bf16* __restrict__ Apack, const __bf16* __restrict__ Bpack,
                      const float* __restrict__ bias, float* __restrict__ C,
                      int Ncols, int rowTiles, int colTiles) {
    int wave = threadIdx.x >> 5;
    int rowTile = blockIdx.x * 8 + wave;
    if (rowTile >= rowTiles) return;
    int lane = threadIdx.x & 31;
    int lo16 = lane & 15, halfL = lane >> 4;

    const v16bf* ap = (const v16bf*)Apack + (long)rowTile * (KT * 32) + lane;
    v16bf a0 = ap[0 * 32];
    v16bf a1 = ap[1 * 32];
    v16bf a2 = ap[2 * 32];
    v16bf a3 = ap[3 * 32];

    for (int ct = 0; ct < colTiles; ++ct) {
        const v16bf* bp = (const v16bf*)Bpack + (long)ct * (KT * 32) + lane;
        // issue all B loads first -> single wait, then chained WMMAs
        v16bf b0 = bp[0 * 32];
        v16bf b1 = bp[1 * 32];
        v16bf b2 = bp[2 * 32];
        v16bf b3 = bp[3 * 32];
        v8f c = {};
        c = wmma_bf16(a0, b0, c);
        c = wmma_bf16(a1, b1, c);
        c = wmma_bf16(a2, b2, c);
        c = wmma_bf16(a3, b3, c);
        int col = ct * 16 + lo16;
        if (col < Ncols) {
            float bv = bias[col];                      // unconditional load
            int rbase = rowTile * 16 + halfL * 8;
#pragma unroll
            for (int r = 0; r < 8; ++r)
                C[(long)(rbase + r) * Ncols + col] = c[r] + bv;
        }
    }
}

// ---------------- mean of edge_attr over E edges (16 features) ----------------
__global__ void ea_reduce(const float* __restrict__ ea, float* __restrict__ easum) {
    __shared__ float s[FE];
    if (threadIdx.x < FE) s[threadIdx.x] = 0.0f;
    __syncthreads();
    float loc[FE];
#pragma unroll
    for (int k = 0; k < FE; ++k) loc[k] = 0.0f;
    int st = gridDim.x * blockDim.x;
    for (int e = blockIdx.x * blockDim.x + threadIdx.x; e < EE; e += st) {
        const float* row = ea + (long)e * FE;
#pragma unroll
        for (int k = 0; k < FE; ++k) loc[k] += row[k];
    }
#pragma unroll
    for (int k = 0; k < FE; ++k) atomicAdd(&s[k], loc[k]);
    __syncthreads();
    if (threadIdx.x < FE) atomicAdd(&easum[threadIdx.x], s[threadIdx.x]);
}

// ---------------- precompute folded edge matrices ----------------
// B[l][k][h] = sum_c W_edge[l][k][h*32+c] * att_edge[l][h][c]   (16x4 per layer)
// loopale[l][h] = sum_k ea_mean[k] * B[l][k][h]                  (self-loop logit)
__global__ void prep_edge(const float* __restrict__ easum,
                          const float* __restrict__ W_edge,
                          const float* __restrict__ att_edge,
                          float* __restrict__ eamean,
                          float* __restrict__ Bmat,
                          float* __restrict__ loopale) {
    __shared__ float sB[128];
    __shared__ float sEm[FE];
    int tid = threadIdx.x;                      // 0..127
    if (tid < FE) {
        float m = easum[tid] * (1.0f / (float)EE);
        eamean[tid] = m;
        sEm[tid] = m;
    }
    int l  = tid >> 6;          // layer
    int k  = (tid >> 2) & 15;   // edge-feature row
    int hh = tid & 3;           // head
    float acc = 0.0f;
#pragma unroll
    for (int c = 0; c < CC; ++c)
        acc += W_edge[l * FE * FEATD + k * FEATD + hh * CC + c] *
               att_edge[l * FEATD + hh * CC + c];
    sB[tid] = acc;
    Bmat[tid] = acc;            // layout: l*64 + k*4 + h == tid
    __syncthreads();
    if (tid < 8) {
        int l2 = tid >> 2, h2 = tid & 3;
        float a2 = 0.0f;
#pragma unroll
        for (int k2 = 0; k2 < FE; ++k2) a2 += sEm[k2] * sB[l2 * 64 + k2 * 4 + h2];
        loopale[tid] = a2;
    }
}

// ---------------- per-node attention scalars ----------------
__global__ void node_alpha(const float* __restrict__ h,
                           const float* __restrict__ a_s,
                           const float* __restrict__ a_d,
                           float* __restrict__ als, float* __restrict__ ald) {
    int idx = blockIdx.x * blockDim.x + threadIdx.x;   // (n, head)
    if (idx >= NN * NH) return;
    int n = idx >> 2, hh = idx & 3;
    const float* hr = h + (long)n * FEATD + hh * CC;
    const float* as = a_s + hh * CC;
    const float* ad = a_d + hh * CC;
    float s = 0.0f, d = 0.0f;
#pragma unroll
    for (int c = 0; c < CC; ++c) { float v = hr[c]; s += v * as[c]; d += v * ad[c]; }
    als[idx] = s;
    ald[idx] = d;
}

// ---------------- edge logits + leaky relu + segment max ----------------
__global__ void edge_logits(const int* __restrict__ ei, const float* __restrict__ ea,
                            const float* __restrict__ Bl, const float* __restrict__ loopale,
                            const float* __restrict__ als, const float* __restrict__ ald,
                            float* __restrict__ exbuf, unsigned* __restrict__ mkey) {
    int gid = blockIdx.x * blockDim.x + threadIdx.x;
    if (gid >= ETOT * NH) return;
    int e = gid >> 2, hh = gid & 3;
    int s, d; float ale;
    if (e < EE) {
        s = ei[e]; d = ei[EE + e];
        const float* row = ea + (long)e * FE;
        float a = 0.0f;
#pragma unroll
        for (int k = 0; k < FE; ++k) a += row[k] * Bl[k * 4 + hh];
        ale = a;
    } else {
        s = d = e - EE;
        ale = loopale[hh];
    }
    float lg = als[s * NH + hh] + ald[d * NH + hh] + ale;
    lg = (lg >= 0.0f) ? lg : NEG_SLOPE * lg;
    exbuf[gid] = lg;                             // stash raw logit
    atomicMax(&mkey[d * NH + hh], fkey(lg));
}

// ---------------- exp(logit - max) + segment sum ----------------
__global__ void edge_exp(const int* __restrict__ ei, float* __restrict__ exbuf,
                         const unsigned* __restrict__ mkey, float* __restrict__ denom) {
    int gid = blockIdx.x * blockDim.x + threadIdx.x;
    if (gid >= ETOT * NH) return;
    int e = gid >> 2, hh = gid & 3;
    int d = (e < EE) ? ei[EE + e] : (e - EE);
    float mx = fkey_inv(mkey[d * NH + hh]);
    float ex = __expf(exbuf[gid] - mx);
    exbuf[gid] = ex;
    atomicAdd(&denom[d * NH + hh], ex);
}

// ---------------- weighted message scatter: agg[dst] += h[src] * alpha ----------------
__global__ __launch_bounds__(256)
void scatter_msg(const int* __restrict__ ei, const float* __restrict__ h,
                 const float* __restrict__ exbuf, const float* __restrict__ denom,
                 float* __restrict__ agg) {
    int gid = blockIdx.x * blockDim.x + threadIdx.x;    // < ETOT*128 (1.09e8)
    int e = gid >> 7;
    int t = gid & 127;
    int hh = t >> 5;
    int s, d;
    if (e < EE) { s = ei[e]; d = ei[EE + e]; }
    else        { s = d = e - EE; }
    float alpha = exbuf[e * NH + hh] / denom[d * NH + hh];
    atomicAdd(&agg[(long)d * FEATD + t], h[(long)s * FEATD + t] * alpha);
}

// ---------------- bias + relu + batch stats ----------------
__global__ __launch_bounds__(256)
void epilogue_stats(const float* __restrict__ agg, const float* __restrict__ bias,
                    float* __restrict__ y, float* __restrict__ bnsum,
                    float* __restrict__ bnsq) {
    __shared__ float ssum[256];
    __shared__ float ssq[256];
    int tid = threadIdx.x;
    int f = tid & 127, rh = tid >> 7;      // 2 row-lanes per feature
    int rbase = blockIdx.x * 32;
    float bv = bias[f];
    float sum = 0.0f, sq = 0.0f;
    for (int i = rh; i < 32; i += 2) {
        int n = rbase + i;
        if (n >= NN) break;
        float v = agg[(long)n * FEATD + f] + bv;
        v = v > 0.0f ? v : 0.0f;
        y[(long)n * FEATD + f] = v;
        sum += v; sq += v * v;
    }
    ssum[tid] = sum; ssq[tid] = sq;
    __syncthreads();
    if (tid < 128) {
        atomicAdd(&bnsum[f], ssum[tid] + ssum[tid + 128]);
        atomicAdd(&bnsq[f],  ssq[tid]  + ssq[tid + 128]);
    }
}

__global__ void bn_final(const float* __restrict__ bnsum, const float* __restrict__ bnsq,
                         const float* __restrict__ gamma, const float* __restrict__ beta,
                         float* __restrict__ bnscale, float* __restrict__ bnshift) {
    int f = threadIdx.x;
    if (f >= FEATD) return;
    float inv = 1.0f / (float)NN;
    float mu = bnsum[f] * inv;
    float var = bnsq[f] * inv - mu * mu;
    var = var > 0.0f ? var : 0.0f;
    float sc = rsqrtf(var + BN_EPS) * gamma[f];
    bnscale[f] = sc;
    bnshift[f] = beta[f] - mu * sc;
}

__global__ void normalize(const float* __restrict__ y, const float* __restrict__ bnscale,
                          const float* __restrict__ bnshift, float* __restrict__ dst) {
    int idx = blockIdx.x * blockDim.x + threadIdx.x;
    if (idx >= NN * FEATD) return;
    int f = idx & 127;
    dst[idx] = y[idx] * bnscale[f] + bnshift[f];
}

// ---------------- launch ----------------
extern "C" void kernel_launch(void* const* d_in, const int* in_sizes, int n_in,
                              void* d_out, int out_size, void* d_ws, size_t ws_size,
                              hipStream_t stream) {
    (void)in_sizes; (void)n_in; (void)out_size; (void)ws_size;
    const float* x        = (const float*)d_in[0];
    const int*   ei       = (const int*)  d_in[1];
    const float* ea       = (const float*)d_in[2];
    const float* W        = (const float*)d_in[3];
    const float* att_src  = (const float*)d_in[4];
    const float* att_dst  = (const float*)d_in[5];
    const float* W_edge   = (const float*)d_in[6];
    const float* att_edge = (const float*)d_in[7];
    const float* bias     = (const float*)d_in[8];
    const float* gamma    = (const float*)d_in[9];
    const float* beta     = (const float*)d_in[10];
    const float* clfW     = (const float*)d_in[11];
    const float* clfb     = (const float*)d_in[12];
    float* out = (float*)d_out;

    const int rowTiles = NN / 16;          // 3125
    const int emboff   = NN * KK;          // embeddings offset in d_out

    // ---- workspace layout ----
    // bf16 packed fragments first (32B aligned), floats after.
    __bf16* Apack = (__bf16*)d_ws;                          // N*128 bf16 (12.8MB)
    __bf16* Bpack = Apack + (long)NN * FEATD;               // 8*KT*32*16 bf16 (32KB)
    float*  wsf   = (float*)(Bpack + 8 * KT * 32 * 16);

    float*    hbuf    = wsf;                                  // N*128
    float*    xbuf    = hbuf + (long)NN * FEATD;              // N*128
    unsigned* mkey    = (unsigned*)(xbuf + (long)NN * FEATD); // N*4   } zeroed
    float*    denom   = (float*)(mkey + (long)NN * NH);       // N*4     } per
    float*    agg     = denom + (long)NN * NH;                // N*128   } layer
    float*    bnsum   = agg + (long)NN * FEATD;               // 128     }
    float*    bnsq    = bnsum + FEATD;                        // 128     }
    float*    als     = bnsq + FEATD;                         // N*4
    float*    ald     = als + (long)NN * NH;                  // N*4
    float*    exbuf   = ald + (long)NN * NH;                  // ETOT*4
    float*    easum   = exbuf + (long)ETOT * NH;              // 16
    float*    eamean  = easum + FE;                           // 16
    float*    Bmat    = eamean + FE;                          // 2*16*4
    float*    loopale = Bmat + 128;                           // 2*4
    float*    bnscale = loopale + 8;                          // 128
    float*    bnshift = bnscale + FEATD;                      // 128
    float*    zbias   = bnshift + FEATD;                      // 128 (zero bias for GEMM)

    const long zeroCount = (long)NN * NH * 2 + (long)NN * FEATD + 2 * FEATD;
    const int  NE4 = ETOT * NH;                               // 3.4M
    const int  packAthreads = rowTiles * KT * 32;             // 400000

    // layer-independent prep: zero bias, edge-attr mean, folded edge matrices
    fill_zero<<<1, 64, 0, stream>>>(easum, FE);
    fill_zero<<<1, 128, 0, stream>>>(zbias, FEATD);
    ea_reduce<<<512, 256, 0, stream>>>(ea, easum);
    prep_edge<<<1, 128, 0, stream>>>(easum, W_edge, att_edge, eamean, Bmat, loopale);

    for (int l = 0; l < 2; ++l) {
        const float* xcur = (l == 0) ? x : xbuf;
        fill_zero<<<2048, 256, 0, stream>>>((float*)mkey, zeroCount);

        // pack activations + weights into WMMA fragment layout (bf16)
        pack_a<<<(packAthreads + 255) / 256, 256, 0, stream>>>(xcur, Apack, rowTiles);
        pack_b<<<(8 * KT * 32 + 255) / 256, 256, 0, stream>>>(
            W + (long)l * FEATD * FEATD, Bpack, FEATD, 8);

        // h = xcur @ W[l]   (pure b128-load + v_wmma inner loop)
        gemm_wmma_packed<<<(rowTiles + 7) / 8, 256, 0, stream>>>(
            Apack, Bpack, zbias, hbuf, FEATD, rowTiles, 8);

        node_alpha<<<(NN * NH + 255) / 256, 256, 0, stream>>>(
            hbuf, att_src + l * FEATD, att_dst + l * FEATD, als, ald);

        edge_logits<<<(NE4 + 255) / 256, 256, 0, stream>>>(
            ei, ea, Bmat + l * 64, loopale + l * 4, als, ald, exbuf, mkey);

        edge_exp<<<(NE4 + 255) / 256, 256, 0, stream>>>(ei, exbuf, mkey, denom);

        scatter_msg<<<(ETOT * 128) / 256, 256, 0, stream>>>(ei, hbuf, exbuf, denom, agg);

        epilogue_stats<<<(NN + 31) / 32, 256, 0, stream>>>(
            agg, bias + l * FEATD, xbuf, bnsum, bnsq);

        bn_final<<<1, 128, 0, stream>>>(bnsum, bnsq, gamma + l * FEATD,
                                        beta + l * FEATD, bnscale, bnshift);

        float* dst = (l == 0) ? xbuf : (out + emboff);     // final layer -> embeddings
        normalize<<<(NN * FEATD + 255) / 256, 256, 0, stream>>>(xbuf, bnscale, bnshift, dst);
    }

    // classifier: out[:, :10] = emb @ clf_W + clf_b
    pack_a<<<(packAthreads + 255) / 256, 256, 0, stream>>>(out + emboff, Apack, rowTiles);
    pack_b<<<1, 256, 0, stream>>>(clfW, Bpack, KK, 1);
    gemm_wmma_packed<<<(rowTiles + 7) / 8, 256, 0, stream>>>(
        Apack, Bpack, clfb, out, KK, rowTiles, 1);
}
